// SelfAttention_44255343018291
// MI455X (gfx1250) — matrix operand order repaired and checked by reading, not compiled, not run on hardware
//
#include <hip/hip_runtime.h>

typedef __attribute__((ext_vector_type(16))) _Float16 v16h;
typedef __attribute__((ext_vector_type(8)))  _Float16 v8h;
typedef __attribute__((ext_vector_type(8)))  float    v8f;

#define DIM      1024
#define HEADS    16
#define HEAD_DIM 64
#define BATCH    4
#define SEQ      2048
#define NTOK     (BATCH * SEQ)      // 8192
#define ATT_SCALE 0.125f            // 64^-0.5

// ---------------------------------------------------------------------------
// WMMA fragment loaders (CDNA5 wave32 layouts, CDNA5 ISA §7.12.2)
// A 16x32 f16: lanes 0-15 -> row=lane, K {0..7,16..23}; lanes 16-31 -> row=lane-16, K {8..15,24..31}
// B 32x16 f16: lanes 0-15 -> col=lane, K {0..7,16..23}; lanes 16-31 -> col=lane-16, K {8..15,24..31}
// C/D 16x16 f32: VGPR r -> row r (lanes 0-15) / r+8 (lanes 16-31), col = lane&15
// ---------------------------------------------------------------------------
__device__ __forceinline__ v16h load_a_frag(const _Float16* p, int ld) {
  const int lane = threadIdx.x & 31;
  const int m    = lane & 15;
  const int koff = (lane >> 4) << 3;          // 0 or 8
  const _Float16* r = p + m * ld + koff;
  v8h lo = *(const v8h*)(r);                  // K = koff .. koff+7
  v8h hi = *(const v8h*)(r + 16);             // K = koff+16 .. koff+23
  v16h a;
#pragma unroll
  for (int i = 0; i < 8; ++i) { a[i] = lo[i]; a[i + 8] = hi[i]; }
  return a;
}

// Loads B[n][k] = src row n, K {koff..koff+7, koff+16..koff+23} (koff = 0 or 8).
// With src = weight rows (row-major along K) this computes X @ W^T.
__device__ __forceinline__ v16h load_b_frag(const _Float16* p, int ld) {
  const int lane = threadIdx.x & 31;
  const int n    = lane & 15;
  const int koff = (lane >> 4) << 3;          // 0 or 8
  const _Float16* r = p + n * ld + koff;
  v8h lo = *(const v8h*)(r);
  v8h hi = *(const v8h*)(r + 16);
  v16h b;
#pragma unroll
  for (int i = 0; i < 8; ++i) { b[i] = lo[i]; b[i + 8] = hi[i]; }
  return b;
}

__device__ __forceinline__ v8f wmma_f16(v16h a, v16h b, v8f c) {
  return __builtin_amdgcn_wmma_f32_16x16x32_f16(false, a, false, b,
                                                (short)0, c, false, false);
}

// One software-pipelined K-step: issue loads for the "fill" fragment set at
// k-offset kld, then run 8 WMMAs on the "consume" set.  Calling this twice
// with swapped sets gives copy-free ping-pong double buffering.
__device__ __forceinline__ void gemm_step(
    const _Float16* arow0, const _Float16* arow1,
    const _Float16* w, int e0, int kld,
    v16h& la0, v16h& la1, v16h (&lb)[4],
    const v16h& ca0, const v16h& ca1, const v16h (&cb)[4],
    v8f (&acc)[2][4]) {
  la0 = load_a_frag(arow0 + kld, DIM);
  la1 = load_a_frag(arow1 + kld, DIM);
#pragma unroll
  for (int t = 0; t < 4; ++t)
    lb[t] = load_b_frag(w + (size_t)(e0 + t * 16) * DIM + kld, DIM);
#pragma unroll
  for (int t = 0; t < 4; ++t) acc[0][t] = wmma_f16(ca0, cb[t], acc[0][t]);
#pragma unroll
  for (int t = 0; t < 4; ++t) acc[1][t] = wmma_f16(ca1, cb[t], acc[1][t]);
}

// ---------------------------------------------------------------------------
// Kernel 0: fp32 -> f16 convert
// ---------------------------------------------------------------------------
__global__ void f32_to_f16_kernel(const float* __restrict__ in,
                                  _Float16* __restrict__ out, int n) {
  int i = blockIdx.x * blockDim.x + threadIdx.x;
  if (i < n) out[i] = (_Float16)in[i];
}

// ---------------------------------------------------------------------------
// Kernel 1: QKV = x @ w_qkv^T, scattered into q[b,h,s,hd] (pre-scaled),
// k[b,h,s,hd], vT[b,h,hd,s].  32x64 tile per wave, ping-pong pipelined.
// ---------------------------------------------------------------------------
__global__ __launch_bounds__(256) void qkv_gemm_kernel(
    const _Float16* __restrict__ xh, const _Float16* __restrict__ wq,
    _Float16* __restrict__ qb, _Float16* __restrict__ kb,
    _Float16* __restrict__ vtb) {
  const int wave = threadIdx.x >> 5;
  const int lane = threadIdx.x & 31;
  const int tile = blockIdx.x * 8 + wave;
  const int ncol = (3 * DIM) / 64;            // 48 column groups of 64
  const int e0   = (tile % ncol) * 64;
  const int t0   = (tile / ncol) * 32;

  const _Float16* arow0 = xh + (size_t)t0 * DIM;
  const _Float16* arow1 = xh + (size_t)(t0 + 16) * DIM;

  v8f acc[2][4] = {{{}, {}, {}, {}}, {{}, {}, {}, {}}};
  v16h aA0, aA1, bA[4], aB0, aB1, bB[4];
  aA0 = load_a_frag(arow0, DIM);
  aA1 = load_a_frag(arow1, DIM);
#pragma unroll
  for (int t = 0; t < 4; ++t)
    bA[t] = load_b_frag(wq + (size_t)(e0 + t * 16) * DIM, DIM);

  for (int k0 = 0; k0 < DIM; k0 += 64) {
    // load set B @ k0+32 while consuming set A @ k0
    gemm_step(arow0, arow1, wq, e0, k0 + 32, aB0, aB1, bB, aA0, aA1, bA, acc);
    // load set A @ k0+64 (wraps dead on last iter) while consuming set B
    gemm_step(arow0, arow1, wq, e0, (k0 + 64) & (DIM - 1),
              aA0, aA1, bA, aB0, aB1, bB, acc);
  }

  const int which = e0 / DIM;                  // 0=q 1=k 2=v (uniform per wave)
  const int h     = (e0 % DIM) / HEAD_DIM;
  const int b_    = t0 / SEQ;
  const int s0    = t0 % SEQ;
  const int n     = lane & 15;
  const int mbase = (lane >> 4) << 3;
#pragma unroll
  for (int mg = 0; mg < 2; ++mg) {
#pragma unroll
    for (int t = 0; t < 4; ++t) {
      const int hd = t * 16 + n;               // e0 is 64-aligned -> hd0 = 0
#pragma unroll
      for (int r = 0; r < 8; ++r) {
        const int s = s0 + mg * 16 + mbase + r;
        float v = acc[mg][t][r];
        if (which == 0) {
          qb[(((size_t)b_ * HEADS + h) * SEQ + s) * HEAD_DIM + hd] =
              (_Float16)(v * ATT_SCALE);
        } else if (which == 1) {
          kb[(((size_t)b_ * HEADS + h) * SEQ + s) * HEAD_DIM + hd] =
              (_Float16)v;
        } else {
          vtb[(((size_t)b_ * HEADS + h) * HEAD_DIM + hd) * SEQ + s] =
              (_Float16)v;
        }
      }
    }
  }
}

// ---------------------------------------------------------------------------
// Kernel 2: flash attention.  One 16-row q tile per wave; 32-key chunks.
// K fragments for the NEXT chunk are reloaded (same registers, no copies)
// right after the score WMMAs consume the current ones; the softmax VALU
// block and P@V WMMAs hide that latency.  V loads are hoisted above softmax.
// ---------------------------------------------------------------------------
__global__ __launch_bounds__(128) void attention_kernel(
    const _Float16* __restrict__ qb, const _Float16* __restrict__ kb,
    const _Float16* __restrict__ vtb, _Float16* __restrict__ ao) {
  __shared__ __align__(16) _Float16 pbuf[4][16 * 32];
  const int wave = threadIdx.x >> 5;
  const int lane = threadIdx.x & 31;
  const int tile = blockIdx.x * 4 + wave;
  const int qtiles = SEQ / 16;                 // 128
  const int bh = tile / qtiles;                // 0..63
  const int q0 = (tile % qtiles) * 16;

  const _Float16* Q  = qb  + ((size_t)bh * SEQ + q0) * HEAD_DIM;
  const _Float16* K  = kb  + (size_t)bh * SEQ * HEAD_DIM;
  const _Float16* VT = vtb + (size_t)bh * HEAD_DIM * SEQ;

  // Q fragments for hd 0..31 and 32..63, held for the whole key loop.
  v16h qa0 = load_a_frag(Q, HEAD_DIM);
  v16h qa1 = load_a_frag(Q + 32, HEAD_DIM);

  v8f o0 = {}, o1 = {}, o2 = {}, o3 = {};
  float mrow[8], lrow[8];
#pragma unroll
  for (int r = 0; r < 8; ++r) { mrow[r] = -3.0e38f; lrow[r] = 0.0f; }

  const int n = lane & 15;
  const int rbase = (lane >> 4) << 3;          // rows rbase..rbase+7 this half
  _Float16* pw = &pbuf[wave][0];

  // Preload K fragments for chunk j = 0.
  v16h kb00 = load_b_frag(K, HEAD_DIM);
  v16h kb01 = load_b_frag(K + 32, HEAD_DIM);
  v16h kb10 = load_b_frag(K + 16 * HEAD_DIM, HEAD_DIM);
  v16h kb11 = load_b_frag(K + 16 * HEAD_DIM + 32, HEAD_DIM);

  for (int j = 0; j < SEQ; j += 32) {
    // S[16x32] = Q[16x64] @ K_chunk^T  (fragments preloaded last iteration)
    v8f s0 = {}, s1 = {};
    s0 = wmma_f16(qa0, kb00, s0);
    s0 = wmma_f16(qa1, kb01, s0);
    s1 = wmma_f16(qa0, kb10, s1);
    s1 = wmma_f16(qa1, kb11, s1);

    // Issue V-chunk fragment loads now; softmax VALU work below hides them.
    v16h vb0 = load_b_frag(VT + (size_t)(0 * 16) * SEQ + j, SEQ);
    v16h vb1 = load_b_frag(VT + (size_t)(1 * 16) * SEQ + j, SEQ);
    v16h vb2 = load_b_frag(VT + (size_t)(2 * 16) * SEQ + j, SEQ);
    v16h vb3 = load_b_frag(VT + (size_t)(3 * 16) * SEQ + j, SEQ);

    // Prefetch next chunk's K fragments into the same registers (no copies;
    // score WMMAs above already consumed the old values).
    const int jn = (j + 32) & (SEQ - 1);       // wraps dead on last iter
    kb00 = load_b_frag(K + (size_t)jn * HEAD_DIM, HEAD_DIM);
    kb01 = load_b_frag(K + (size_t)jn * HEAD_DIM + 32, HEAD_DIM);
    kb10 = load_b_frag(K + (size_t)(jn + 16) * HEAD_DIM, HEAD_DIM);
    kb11 = load_b_frag(K + (size_t)(jn + 16) * HEAD_DIM + 32, HEAD_DIM);
    // L2 prefetch of the K chunk two iterations ahead: 32 lanes x 128B
    // covers the whole 4KB chunk (lowers to global_prefetch_b8).
    const int j2 = (j + 64) & (SEQ - 1);
    __builtin_prefetch(K + (size_t)(j2 + lane) * HEAD_DIM, 0, 3);

    // Online softmax: row r of this half lives across lanes of the same
    // 16-lane half -> xor-shuffles with masks 1,2,4,8 reduce the row.
#pragma unroll
    for (int r = 0; r < 8; ++r) {
      float smax = fmaxf(s0[r], s1[r]);
#pragma unroll
      for (int off = 1; off < 16; off <<= 1)
        smax = fmaxf(smax, __shfl_xor(smax, off, 32));
      float mnew  = fmaxf(mrow[r], smax);
      float alpha = __expf(mrow[r] - mnew);
      float p0 = __expf(s0[r] - mnew);
      float p1 = __expf(s1[r] - mnew);
      float psum = p0 + p1;
#pragma unroll
      for (int off = 1; off < 16; off <<= 1)
        psum += __shfl_xor(psum, off, 32);
      lrow[r] = lrow[r] * alpha + psum;
      mrow[r] = mnew;
      o0[r] = o0[r] * alpha; o1[r] = o1[r] * alpha;
      o2[r] = o2[r] * alpha; o3[r] = o3[r] * alpha;
      // Stash P (f16) to LDS in row-major [16][32] for A-fragment reload.
      const int row = r + rbase;
      pw[row * 32 + n]      = (_Float16)p0;
      pw[row * 32 + n + 16] = (_Float16)p1;
    }
    // O += P[16x32] @ V_chunk[32x64]  (V stored transposed: contiguous in s)
    v16h pa = load_a_frag(pw, 32);             // same-wave DS ops are in-order
    o0 = wmma_f16(pa, vb0, o0);
    o1 = wmma_f16(pa, vb1, o1);
    o2 = wmma_f16(pa, vb2, o2);
    o3 = wmma_f16(pa, vb3, o3);
  }

  // Normalize and store to [b][s][h*64+hd] (f16) for the output GEMM.
  const int b_ = bh / HEADS, h = bh % HEADS;
#pragma unroll
  for (int r = 0; r < 8; ++r) {
    float rl = 1.0f / lrow[r];
    const size_t base =
        ((size_t)b_ * SEQ + (q0 + rbase + r)) * DIM + h * HEAD_DIM + n;
    ao[base +  0] = (_Float16)(o0[r] * rl);
    ao[base + 16] = (_Float16)(o1[r] * rl);
    ao[base + 32] = (_Float16)(o2[r] * rl);
    ao[base + 48] = (_Float16)(o3[r] * rl);
  }
}

// ---------------------------------------------------------------------------
// Kernel 3: out = attn_out @ w_out^T + b_out  (fp32 result)
// 32x64 tile per wave, ping-pong pipelined like the QKV GEMM.
// ---------------------------------------------------------------------------
__global__ __launch_bounds__(256) void out_gemm_kernel(
    const _Float16* __restrict__ ao, const _Float16* __restrict__ wo,
    const float* __restrict__ bias, float* __restrict__ out) {
  const int wave = threadIdx.x >> 5;
  const int lane = threadIdx.x & 31;
  const int tile = blockIdx.x * 8 + wave;
  const int ncol = DIM / 64;                  // 16 column groups of 64
  const int e0   = (tile % ncol) * 64;
  const int t0   = (tile / ncol) * 32;

  const _Float16* arow0 = ao + (size_t)t0 * DIM;
  const _Float16* arow1 = ao + (size_t)(t0 + 16) * DIM;

  v8f acc[2][4] = {{{}, {}, {}, {}}, {{}, {}, {}, {}}};
  v16h aA0, aA1, bA[4], aB0, aB1, bB[4];
  aA0 = load_a_frag(arow0, DIM);
  aA1 = load_a_frag(arow1, DIM);
#pragma unroll
  for (int t = 0; t < 4; ++t)
    bA[t] = load_b_frag(wo + (size_t)(e0 + t * 16) * DIM, DIM);

  for (int k0 = 0; k0 < DIM; k0 += 64) {
    gemm_step(arow0, arow1, wo, e0, k0 + 32, aB0, aB1, bB, aA0, aA1, bA, acc);
    gemm_step(arow0, arow1, wo, e0, (k0 + 64) & (DIM - 1),
              aA0, aA1, bA, aB0, aB1, bB, acc);
  }

  const int n     = lane & 15;
  const int mbase = (lane >> 4) << 3;
#pragma unroll
  for (int mg = 0; mg < 2; ++mg) {
#pragma unroll
    for (int t = 0; t < 4; ++t) {
      const float bv = bias[e0 + t * 16 + n];
#pragma unroll
      for (int r = 0; r < 8; ++r) {
        out[(size_t)(t0 + mg * 16 + mbase + r) * DIM + e0 + t * 16 + n] =
            acc[mg][t][r] + bv;
      }
    }
  }
}

// ---------------------------------------------------------------------------
extern "C" void kernel_launch(void* const* d_in, const int* in_sizes, int n_in,
                              void* d_out, int out_size, void* d_ws,
                              size_t ws_size, hipStream_t stream) {
  (void)in_sizes; (void)n_in; (void)out_size; (void)ws_size;
  const float* x     = (const float*)d_in[0];
  const float* w_qkv = (const float*)d_in[1];
  const float* w_out = (const float*)d_in[2];
  const float* b_out = (const float*)d_in[3];
  float* out = (float*)d_out;

  _Float16* ws  = (_Float16*)d_ws;
  _Float16* xh  = ws;                                   // [8192,1024]
  _Float16* wqh = xh  + (size_t)NTOK * DIM;             // [3072,1024]
  _Float16* woh = wqh + (size_t)3 * DIM * DIM;          // [1024,1024]
  _Float16* qb  = woh + (size_t)DIM * DIM;              // [4,16,2048,64]
  _Float16* kb  = qb  + (size_t)NTOK * DIM;             // [4,16,2048,64]
  _Float16* vtb = kb  + (size_t)NTOK * DIM;             // [4,16,64,2048]
  _Float16* ao  = vtb + (size_t)NTOK * DIM;             // [8192,1024]

  int n = NTOK * DIM;
  f32_to_f16_kernel<<<(n + 255) / 256, 256, 0, stream>>>(x, xh, n);
  n = 3 * DIM * DIM;
  f32_to_f16_kernel<<<(n + 255) / 256, 256, 0, stream>>>(w_qkv, wqh, n);
  n = DIM * DIM;
  f32_to_f16_kernel<<<(n + 255) / 256, 256, 0, stream>>>(w_out, woh, n);

  {
    int tiles = (NTOK / 32) * ((3 * DIM) / 64);  // 256 * 48 = 12288
    qkv_gemm_kernel<<<tiles / 8, 256, 0, stream>>>(xh, wqh, qb, kb, vtb);
  }
  {
    int qtiles = BATCH * HEADS * (SEQ / 16);     // 8192
    attention_kernel<<<qtiles / 4, 128, 0, stream>>>(qb, kb, vtb, ao);
  }
  {
    int tiles = (NTOK / 32) * (DIM / 64);        // 256 * 16 = 4096
    out_gemm_kernel<<<tiles / 8, 256, 0, stream>>>(ao, woh, b_out, out);
  }
}
